// Net0_61014305407684
// MI455X (gfx1250) — compile-verified, matrix operand order checked by reading, not run on hardware
//
#include <hip/hip_runtime.h>
#include <hip/hip_bf16.h>

typedef __attribute__((ext_vector_type(16))) _Float16 v16h;
typedef __attribute__((ext_vector_type(8)))  float    v8f;
typedef __attribute__((ext_vector_type(2)))  __fp16   pk2;  // return type of cvt_pkrtz

#define NLAYER 21   // fc1 + 19 mid + fc21
#define TILES  4    // 16-row batch tiles per wave -> 64 rows/wave

union AFrag { v16h h; uint32_t u[8]; };

static __device__ __forceinline__ uint32_t pkh(float a, float b) {
    pk2 h = __builtin_amdgcn_cvt_pkrtz(a, b);   // v_cvt_pk_f16_f32 (RTZ: never makes Inf)
    union { pk2 h; uint32_t u; } c; c.h = h; return c.u;
}

// Swap data between lane i and lane i^16: ds_swizzle_b32 SWAPX16
// group-of-32 encoding: offset = xor_mask<<10 | or_mask<<5 | and_mask = 0x401F
static __device__ __forceinline__ float swap16(float v) {
    int i = __builtin_bit_cast(int, v);
    i = __builtin_amdgcn_ds_swizzle(i, 0x401F);
    return __builtin_bit_cast(float, i);
}

// branchless ELU: x>0 -> x ; x<=0 -> exp(x)-1
static __device__ __forceinline__ float elu(float x) {
    return fmaxf(x, 0.f) + (__expf(fminf(x, 0.f)) - 1.f);
}

__global__ __launch_bounds__(256) void mlp21_wmma_kernel(
    const float* __restrict__ x,    const float* __restrict__ W1,
    const float* __restrict__ b1,   const float* __restrict__ Wmid,
    const float* __restrict__ bmid, const float* __restrict__ W21,
    const float* __restrict__ b21,  float* __restrict__ out, int nrows)
{
    // Weights stored in LDS pre-swizzled into WMMA A-fragment rows:
    // sW[layer][r][0..7]  = W[r][k=0..7]  (lane r,    VGPR0..3)
    // sW[layer][r][8..15] = W[r][k=8..15] (lane r+16, VGPR0..3), zero padded
    __shared__ __align__(16) _Float16 sW[NLAYER][16][16];
    __shared__ __align__(16) float    sB[NLAYER][16];

    for (int i = threadIdx.x; i < NLAYER * 256; i += blockDim.x) {
        int layer = i >> 8, r = (i >> 4) & 15, k = i & 15;
        float w = 0.f;
        if (layer == 0)      { if (r < 9 && k < 2) w = W1[r * 2 + k]; }
        else if (layer < 20) { if (r < 9 && k < 9) w = Wmid[(layer - 1) * 81 + r * 9 + k]; }
        else                 { if (r < 2 && k < 9) w = W21[r * 9 + k]; }
        sW[layer][r][k] = (_Float16)w;
    }
    for (int i = threadIdx.x; i < NLAYER * 16; i += blockDim.x) {
        int layer = i >> 4, r = i & 15;
        float b = 0.f;
        if (layer == 0)      { if (r < 9) b = b1[r]; }
        else if (layer < 20) { if (r < 9) b = bmid[(layer - 1) * 9 + r]; }
        else                 { if (r < 2) b = b21[r]; }
        sB[layer][r] = b;
    }
    __syncthreads();

    const int  lane = threadIdx.x & 31;
    const int  wave = threadIdx.x >> 5;
    const long gw   = (long)blockIdx.x * (blockDim.x >> 5) + wave;
    const long base = gw * (TILES * 16);
    const int  r    = lane & 15;
    const int  hi   = lane >> 4;

    // ---- A fragment (weights) from LDS: one ds_load_b128 per lane ----
    auto load_a = [&](int layer) -> v16h {
        AFrag a;
        const uint32_t* p = (const uint32_t*)&sW[layer][r][hi * 8];
        a.u[0] = p[0]; a.u[1] = p[1]; a.u[2] = p[2]; a.u[3] = p[3];
        a.u[4] = 0;    a.u[5] = 0;    a.u[6] = 0;    a.u[7] = 0;  // K=16..31 pad
        return a.h;
    };
    // ---- bias as WMMA C operand (free bias add) ----
    auto load_c = [&](int layer) -> v8f {
        v8f c;
        const float* bp = &sB[layer][hi * 8];
        #pragma unroll
        for (int j = 0; j < 8; ++j) c[j] = bp[j];
        return c;
    };
    // ---- D (lane=batch col, features in VGPRs) -> next-layer B fragment ----
    // Elements 8..15 (K=16..31) carry finite junk: A is zero-padded over K=16..31,
    // and cvt_pkrtz can't produce Inf, so junk * 0 == 0 -> no masking needed.
    auto build_b = [&](const v8f& d) -> v16h {
        float e0 = elu(d[0]), e1 = elu(d[1]), e2 = elu(d[2]), e3 = elu(d[3]);
        float e4 = elu(d[4]), e5 = elu(d[5]), e6 = elu(d[6]), e7 = elu(d[7]);
        // feature 8 lives in lanes 16..31 (their V0); swap halves (1 ds_swizzle)
        float h8 = swap16(e0);
        AFrag b;
        b.u[0] = pkh(e0, e1);    // K=0,1   (lo lanes; hi lanes: K=8,9)
        b.u[1] = pkh(e2, e3);    // K=2,3
        b.u[2] = pkh(e4, e5);    // K=4,5
        b.u[3] = pkh(e6, e7);    // K=6,7
        b.u[4] = pkh(h8, 0.f);   // K=16,17 / 24,25 (A=0 there)
        b.u[5] = 0; b.u[6] = 0; b.u[7] = 0;
        return b.h;
    };

    // ---- layer 0 input: B = x^T (K=0,1 = x features, N = 16 batch rows) ----
    const float2* xp = (const float2*)x;
    v16h bfrag[TILES];
    #pragma unroll
    for (int t = 0; t < TILES; ++t) {
        long row = base + (long)t * 16 + r;
        float xa = 0.f, xb = 0.f;
        if (row < nrows) { float2 v = xp[row]; xa = v.x; xb = v.y; }
        AFrag b;
        b.u[0] = pkh(xa, xb);    // hi-lane copy lands on K=8,9: A is 0 there
        b.u[1] = 0; b.u[2] = 0; b.u[3] = 0;
        b.u[4] = 0; b.u[5] = 0; b.u[6] = 0; b.u[7] = 0;
        bfrag[t] = b.h;
    }

    v8f acc[TILES];

    // Fully-unrolled, software-pipelined layer chain:
    // issue next layer's LDS weight/bias loads before this layer's WMMAs,
    // so ds latency hides under XDL + ELU(TRANS) work.
    v16h a = load_a(0);
    v8f  c = load_c(0);
    #pragma unroll
    for (int layer = 0; layer < NLAYER; ++layer) {
        v16h an;
        v8f  cn;
        if (layer < NLAYER - 1) { an = load_a(layer + 1); cn = load_c(layer + 1); }
        #pragma unroll
        for (int t = 0; t < TILES; ++t)
            acc[t] = __builtin_amdgcn_wmma_f32_16x16x32_f16(
                false, a, false, bfrag[t], (short)0, c, false, false);
        if (layer < NLAYER - 1) {
            #pragma unroll
            for (int t = 0; t < TILES; ++t) bfrag[t] = build_b(acc[t]);
            a = an; c = cn;
        }
    }

    // log_softmax over 2 logits; logits for batch col n sit in lane n (V0,V1)
    float2* op = (float2*)out;
    if (lane < 16) {
        #pragma unroll
        for (int t = 0; t < TILES; ++t) {
            long row = base + (long)t * 16 + r;
            if (row < nrows) {
                float l0 = acc[t][0], l1 = acc[t][1];
                float mx = fmaxf(l0, l1);
                float ls = mx + __logf(__expf(l0 - mx) + __expf(l1 - mx));
                float2 o; o.x = l0 - ls; o.y = l1 - ls;
                op[row] = o;
            }
        }
    }
}

extern "C" void kernel_launch(void* const* d_in, const int* in_sizes, int n_in,
                              void* d_out, int out_size, void* d_ws, size_t ws_size,
                              hipStream_t stream) {
    const float* x    = (const float*)d_in[0];
    const float* W1   = (const float*)d_in[1];
    const float* b1   = (const float*)d_in[2];
    const float* Wmid = (const float*)d_in[3];
    const float* bmid = (const float*)d_in[4];
    const float* W21  = (const float*)d_in[5];
    const float* b21  = (const float*)d_in[6];
    float* out = (float*)d_out;

    int nrows = in_sizes[0] / 2;                    // x is [B,2]
    int rows_per_block = (256 / 32) * (TILES * 16); // 8 waves * 64 rows = 512
    int grid = (nrows + rows_per_block - 1) / rows_per_block;

    mlp21_wmma_kernel<<<grid, 256, 0, stream>>>(x, W1, b1, Wmid, bmid, W21, b21,
                                                out, nrows);
}
